// GraphTransformerLayer_43465069035641
// MI455X (gfx1250) — matrix operand order repaired and checked by reading, not compiled, not run on hardware
//
#include <hip/hip_runtime.h>
#include <math.h>

typedef __attribute__((ext_vector_type(16))) _Float16 v16h;
typedef __attribute__((ext_vector_type(8)))  _Float16 v8h;
typedef __attribute__((ext_vector_type(8)))  float    v8f;
typedef __attribute__((ext_vector_type(4)))  unsigned v4u;
typedef __attribute__((ext_vector_type(8)))  int      v8i;
typedef __attribute__((ext_vector_type(4)))  int      v4i;

#define DIM 256
#define NH  8
#define HD  32
#define NEG_SLOPE 0.2f
#define APAD 8                    // 8 halfwords = 16B pad per LDS row
#define ASTR (DIM + APAD)         // 264 halfwords = 528B row stride

// ---- monotone float<->uint encoding for atomicMax on signed floats ----
__device__ __forceinline__ unsigned fenc(float f) {
  unsigned u = __float_as_uint(f);
  return (u & 0x80000000u) ? ~u : (u | 0x80000000u);
}
__device__ __forceinline__ float fdec(unsigned e) {
  unsigned b = (e & 0x80000000u) ? (e & 0x7FFFFFFFu) : ~e;
  return __uint_as_float(b);
}

// ======================================================================
// Precision prep: f32 -> f16 copies, and f16 transposed weights
// ======================================================================
__global__ void f32_to_f16(const float* __restrict__ in,
                           _Float16* __restrict__ out, int n) {
  int i = blockIdx.x * blockDim.x + threadIdx.x;
  if (i < n) out[i] = (_Float16)in[i];
}

// Wt[n][k] = (f16) W[k][n]   (256x256)
__global__ __launch_bounds__(256)
void transpose_to_f16(const float* __restrict__ W, _Float16* __restrict__ Wt) {
  int n = blockIdx.x;
  int k = threadIdx.x;
  Wt[n * DIM + k] = (_Float16)W[(size_t)k * DIM + n];
}

// ======================================================================
// WMMA GEMM: out[N,256] = act(A[N,256](f16) @ Wt^T + bias)
//   Wt is [256 cols][256 k] (transposed, f16) so B fragments are two
//   contiguous b128 loads per lane.
// Block = 256 threads = 8 waves, covers 16 rows x 256 cols.
// A tile (16x256 f16, 8KB) is DMA'd into LDS by the Tensor Data Mover,
// with TDM padding of 4 dwords per 128-dword row -> 528B LDS row stride
// (bank-conflict-free fragment reads). Waves read A via ds_load_b128.
// ======================================================================
template <int ACT, int OUTH>
__global__ __launch_bounds__(256)
void gemm256_wmma(const _Float16* __restrict__ A, const _Float16* __restrict__ Bt,
                  const float* __restrict__ bias, void* __restrict__ outv,
                  int nrows) {
  __shared__ _Float16 atile[16 * ASTR];

  const int lane = threadIdx.x & 31;
  const int wave = threadIdx.x >> 5;
  const int row0 = blockIdx.x * 16;

  if (wave == 0) {
    // --- Tensor DMA descriptor (D#) ---
    unsigned long long ga =
        (unsigned long long)(const void*)(A + (size_t)row0 * DIM);
    unsigned lds_off = (unsigned)(unsigned long long)(const void*)atile;

    v4u g0;
    g0[0] = 1u;                                   // count=1, user mode
    g0[1] = lds_off;                              // lds_addr
    g0[2] = (unsigned)ga;                         // global_addr[31:0]
    g0[3] = (unsigned)((ga >> 32) & 0x01FFFFFFull) | (2u << 30); // [56:32], type=2

    v8i g1;
    // data_size=2B, pad_enable, pad_interval=128dw (one row), pad_amount=4dw
    g1[0] = (int)((1u << 16) | (1u << 20) | (6u << 22) | (3u << 25));
    g1[1] = (int)(((unsigned)DIM & 0xFFFFu) << 16);      // tensor_dim0 lo
    g1[2] = (int)(((unsigned)nrows & 0xFFFFu) << 16);    // dim0 hi=0 | dim1 lo
    g1[3] = (int)((((unsigned)nrows >> 16) & 0xFFFFu) | ((unsigned)DIM << 16)); // dim1 hi | tile_dim0=256
    g1[4] = 16;                                          // tile_dim1=16
    g1[5] = DIM;                                         // tensor_dim0_stride lo
    g1[6] = 0;
    g1[7] = 0;
    v4i gz = {0, 0, 0, 0};
#if __clang_major__ >= 23
    v8i gz8 = {0, 0, 0, 0, 0, 0, 0, 0};
    __builtin_amdgcn_tensor_load_to_lds(g0, g1, gz, gz, gz8, 0);
#else
    __builtin_amdgcn_tensor_load_to_lds(g0, g1, gz, gz, 0);
#endif
    __builtin_amdgcn_s_wait_tensorcnt(0);
  }
  __syncthreads();

  const int m    = lane & 15;
  const int half = lane >> 4;
  const int col0 = wave * 32;

  const _Float16* arow  = &atile[m * ASTR];
  const _Float16* bcol0 = Bt + (size_t)(col0 + m) * DIM;
  const _Float16* bcol1 = bcol0 + 16 * DIM;

  v8f c0 = {};
  v8f c1 = {};

#pragma unroll
  for (int k0 = 0; k0 < DIM; k0 += 32) {
    // A fragment (16-bit 16x32 layout): elems 0..7 -> K=k0+8*half+{0..7},
    // elems 8..15 -> K=k0+8*half+16+{0..7}
    const int ka = k0 + half * 8;
    v8h a_lo = *(const v8h*)(arow + ka);
    v8h a_hi = *(const v8h*)(arow + ka + 16);
    v16h af;
#pragma unroll
    for (int i = 0; i < 8; ++i) { af[i] = a_lo[i]; af[i + 8] = a_hi[i]; }

    // B fragment: lane=col, elems 0..7 -> K = k0 + 8*half + j, elems 8..15 -> K = k0 + 8*half + 16 + j
    const int kb = k0 + half * 8;
    v8h b0lo = *(const v8h*)(bcol0 + kb);
    v8h b0hi = *(const v8h*)(bcol0 + kb + 16);
    v8h b1lo = *(const v8h*)(bcol1 + kb);
    v8h b1hi = *(const v8h*)(bcol1 + kb + 16);
    v16h b0f, b1f;
#pragma unroll
    for (int j = 0; j < 8; ++j) {
      b0f[j] = b0lo[j]; b0f[j + 8] = b0hi[j];
      b1f[j] = b1lo[j]; b1f[j + 8] = b1hi[j];
    }

    c0 = __builtin_amdgcn_wmma_f32_16x16x32_f16(false, af, false, b0f,
                                                (short)0, c0, false, false);
    c1 = __builtin_amdgcn_wmma_f32_16x16x32_f16(false, af, false, b1f,
                                                (short)0, c1, false, false);
  }

  // D layout: elem r -> row = row0 + r + 8*half, col = tile_col + (lane&15)
#pragma unroll
  for (int r = 0; r < 8; ++r) {
    const int row = row0 + r + half * 8;
    const int cA  = col0 + m;
    float v0 = c0[r], v1 = c1[r];
    if (bias) { v0 += bias[cA]; v1 += bias[cA + 16]; }
    if (ACT == 1) { v0 = fmaxf(v0, 0.f); v1 = fmaxf(v1, 0.f); }
    if (OUTH) {
      _Float16* o = (_Float16*)outv;
      o[(size_t)row * DIM + cA]      = (_Float16)v0;
      o[(size_t)row * DIM + cA + 16] = (_Float16)v1;
    } else {
      float* o = (float*)outv;
      o[(size_t)row * DIM + cA]      = v0;
      o[(size_t)row * DIM + cA + 16] = v1;
    }
  }
}

// ======================================================================
// per-(node,head) attention logits
// ======================================================================
__global__ void att_scores(const float* __restrict__ xp,
                           const float* __restrict__ att_src,
                           const float* __restrict__ att_dst,
                           float* __restrict__ a_src, float* __restrict__ a_dst,
                           int n) {
  int idx = blockIdx.x * blockDim.x + threadIdx.x;
  if (idx >= n * NH) return;
  int node = idx >> 3, h = idx & 7;
  const float* xr = xp + (size_t)node * DIM + h * HD;
  const float* as = att_src + h * HD;
  const float* ad = att_dst + h * HD;
  float s = 0.f, d = 0.f;
#pragma unroll
  for (int i = 0; i < HD; ++i) { float v = xr[i]; s += v * as[i]; d += v * ad[i]; }
  a_src[idx] = s;
  a_dst[idx] = d;
}

__global__ void init_ws(float* __restrict__ accum, unsigned* __restrict__ m_enc,
                        float* __restrict__ denom, int n) {
  int i = blockIdx.x * blockDim.x + threadIdx.x;
  if (i < n * DIM) accum[i] = 0.f;
  if (i < n * NH) { m_enc[i] = fenc(-INFINITY); denom[i] = 0.f; }
}

__device__ __forceinline__ void edge_nodes(const int* __restrict__ ei, int e,
                                           int Ereal, int& src, int& dst) {
  if (e < Ereal) { src = ei[e]; dst = ei[Ereal + e]; }
  else           { src = e - Ereal; dst = e - Ereal; }  // self loop
}

// pass 1: segment max over destinations
__global__ void edge_max(const int* __restrict__ ei,
                         const float* __restrict__ a_src,
                         const float* __restrict__ a_dst,
                         unsigned* __restrict__ m_enc, int Etot, int Ereal) {
  int e = blockIdx.x * blockDim.x + threadIdx.x;
  if (e >= Etot) return;
  int src, dst; edge_nodes(ei, e, Ereal, src, dst);
#pragma unroll
  for (int h = 0; h < NH; ++h) {
    float v = a_src[src * NH + h] + a_dst[dst * NH + h];
    v = (v > 0.f) ? v : NEG_SLOPE * v;
    atomicMax(&m_enc[dst * NH + h], fenc(v));
  }
}

// pass 2: softmax denominators
__global__ void edge_sum(const int* __restrict__ ei,
                         const float* __restrict__ a_src,
                         const float* __restrict__ a_dst,
                         const unsigned* __restrict__ m_enc,
                         float* __restrict__ denom, int Etot, int Ereal) {
  int e = blockIdx.x * blockDim.x + threadIdx.x;
  if (e >= Etot) return;
  int src, dst; edge_nodes(ei, e, Ereal, src, dst);
#pragma unroll
  for (int h = 0; h < NH; ++h) {
    float v = a_src[src * NH + h] + a_dst[dst * NH + h];
    v = (v > 0.f) ? v : NEG_SLOPE * v;
    float ex = __expf(v - fdec(m_enc[dst * NH + h]));
    atomicAdd(&denom[dst * NH + h], ex);
  }
}

// pass 3: one wave32 per edge; lane handles feature d = h*32 + lane per iter
__global__ __launch_bounds__(256)
void edge_scatter(const int* __restrict__ ei, const float* __restrict__ a_src,
                  const float* __restrict__ a_dst,
                  const unsigned* __restrict__ m_enc,
                  const float* __restrict__ denom, const float* __restrict__ xp,
                  float* __restrict__ accum, int Etot, int Ereal) {
  int g    = (blockIdx.x * blockDim.x + threadIdx.x) >> 5;
  int lane = threadIdx.x & 31;
  if (g >= Etot) return;
  int src, dst; edge_nodes(ei, g, Ereal, src, dst);
  const float* xr = xp + (size_t)src * DIM;
  float* ar = accum + (size_t)dst * DIM;
  __builtin_prefetch(xr + lane, 0, 3);   // global_prefetch_b8 of gather row
  __builtin_prefetch(ar + lane, 1, 3);
#pragma unroll
  for (int h = 0; h < NH; ++h) {
    float v = a_src[src * NH + h] + a_dst[dst * NH + h];
    v = (v > 0.f) ? v : NEG_SLOPE * v;
    float ex    = __expf(v - fdec(m_enc[dst * NH + h]));
    float alpha = ex / (denom[dst * NH + h] + 1e-16f);
    int d = h * HD + lane;
    atomicAdd(&ar[d], alpha * xr[d]);
  }
}

// ======================================================================
// bias + residual + LayerNorm; writes f16 hn (input to FFN GEMM)
// ======================================================================
__global__ __launch_bounds__(256)
void bias_res_ln(const float* __restrict__ accum, const float* __restrict__ x,
                 const float* __restrict__ bias, const float* __restrict__ g,
                 const float* __restrict__ b, _Float16* __restrict__ hn, int n) {
  int row  = (blockIdx.x * blockDim.x + threadIdx.x) >> 5;
  int lane = threadIdx.x & 31;
  if (row >= n) return;
  const float* ar = accum + (size_t)row * DIM;
  const float* xr = x + (size_t)row * DIM;
  float rv[8];
  float s = 0.f;
#pragma unroll
  for (int i = 0; i < 8; ++i) {
    int d = i * 32 + lane;
    rv[i] = ar[d] + bias[d] + xr[d];
    s += rv[i];
  }
#pragma unroll
  for (int o = 16; o > 0; o >>= 1) s += __shfl_xor(s, o, 32);
  float mu = s * (1.f / DIM);
  float vs = 0.f;
#pragma unroll
  for (int i = 0; i < 8; ++i) { float dv = rv[i] - mu; vs += dv * dv; }
#pragma unroll
  for (int o = 16; o > 0; o >>= 1) vs += __shfl_xor(vs, o, 32);
  float rstd = rsqrtf(vs * (1.f / DIM) + 1e-5f);
  _Float16* hr = hn + (size_t)row * DIM;
#pragma unroll
  for (int i = 0; i < 8; ++i) {
    int d = i * 32 + lane;
    hr[d] = (_Float16)((rv[i] - mu) * rstd * g[d] + b[d]);
  }
}

extern "C" void kernel_launch(void* const* d_in, const int* in_sizes, int n_in,
                              void* d_out, int out_size, void* d_ws, size_t ws_size,
                              hipStream_t stream) {
  const float* x    = (const float*)d_in[0];
  const int*   ei   = (const int*)d_in[1];
  const float* W    = (const float*)d_in[3];
  const float* atts = (const float*)d_in[4];
  const float* attd = (const float*)d_in[5];
  const float* bias = (const float*)d_in[6];
  const float* ln_g = (const float*)d_in[7];
  const float* ln_b = (const float*)d_in[8];
  const float* W1   = (const float*)d_in[9];
  const float* b1   = (const float*)d_in[10];
  const float* W2   = (const float*)d_in[11];
  const float* b2   = (const float*)d_in[12];
  float* out = (float*)d_out;

  const int N = in_sizes[0] / DIM;  // 50000 (divisible by 16)
  const int E = in_sizes[1] / 2;    // 800000
  const int T = E + N;

  // workspace layout:
  // xp f32 | accum f32 (reused as th f16) | a_src | a_dst | m_enc | denom |
  // xh f16 (reused as hn f16) | Wt f16 | W1t f16 | W2t f16
  float*    xp    = (float*)d_ws;
  float*    accum = xp + (size_t)N * DIM;
  float*    a_src = accum + (size_t)N * DIM;
  float*    a_dst = a_src + (size_t)N * NH;
  unsigned* m_enc = (unsigned*)(a_dst + (size_t)N * NH);
  float*    denom = (float*)(m_enc + (size_t)N * NH);
  _Float16* xh    = (_Float16*)(denom + (size_t)N * NH);
  _Float16* Wt    = xh + (size_t)N * DIM;
  _Float16* W1t   = Wt + DIM * DIM;
  _Float16* W2t   = W1t + DIM * DIM;
  _Float16* th    = (_Float16*)accum;   // FFN mid, reuses accum after LN

  // 0) precision prep: f16 activations + transposed f16 weights
  f32_to_f16<<<(N * DIM + 255) / 256, 256, 0, stream>>>(x, xh, N * DIM);
  transpose_to_f16<<<DIM, 256, 0, stream>>>(W, Wt);
  transpose_to_f16<<<DIM, 256, 0, stream>>>(W1, W1t);
  transpose_to_f16<<<DIM, 256, 0, stream>>>(W2, W2t);

  // 1) xp = x @ W  (WMMA, TDM-staged A tiles)
  gemm256_wmma<0, 0><<<N / 16, 256, 0, stream>>>(xh, Wt, nullptr, xp, N);
  // 2) per-(node,head) logits
  att_scores<<<(N * NH + 255) / 256, 256, 0, stream>>>(xp, atts, attd, a_src, a_dst, N);
  // 3) init accum / max / denom
  init_ws<<<(N * DIM + 255) / 256, 256, 0, stream>>>(accum, m_enc, denom, N);
  // 4) segment max
  edge_max<<<(T + 255) / 256, 256, 0, stream>>>(ei, a_src, a_dst, m_enc, T, E);
  // 5) softmax denominators
  edge_sum<<<(T + 255) / 256, 256, 0, stream>>>(ei, a_src, a_dst, m_enc, denom, T, E);
  // 6) weighted message scatter (wave per edge)
  edge_scatter<<<(int)(((size_t)T * 32 + 255) / 256), 256, 0, stream>>>(
      ei, a_src, a_dst, m_enc, denom, xp, accum, T, E);
  // 7) bias + residual + LayerNorm -> hn (f16, reuses xh buffer)
  bias_res_ln<<<(int)(((size_t)N * 32 + 255) / 256), 256, 0, stream>>>(
      accum, x, bias, ln_g, ln_b, xh, N);
  // 8) FFN: relu(hn@W1+b1) -> th (f16, in accum buffer); then @W2+b2 -> out
  gemm256_wmma<1, 1><<<N / 16, 256, 0, stream>>>(xh, W1t, b1, th, N);
  gemm256_wmma<0, 0><<<N / 16, 256, 0, stream>>>(th, W2t, b2, out, N);
}
